// AlibiAttention_33509334844056
// MI455X (gfx1250) — compile-verified, matrix operand order checked by reading, not run on hardware
//
#include <hip/hip_runtime.h>
#include <hip/hip_bf16.h>

typedef __attribute__((ext_vector_type(16))) _Float16 v16h;
typedef __attribute__((ext_vector_type(8)))  _Float16 v8h;
typedef __attribute__((ext_vector_type(4)))  _Float16 v4h;
typedef __attribute__((ext_vector_type(8)))  float    v8f;

#define WMMA_F16(a, b, c) \
    __builtin_amdgcn_wmma_f32_16x16x32_f16(false, (a), false, (b), (short)0, (c), false, false)

// ---------------------------------------------------------------------------
// Fragment loader for V_WMMA_F32_16X16X32_F16 A/B operands.
// Source is row-major [16 x >=32] f16 with leading dim `ld` (halves).
// Per ISA 7.12.2 (16-bit A 16x32): lane L holds row M = L&15; element e holds
// K = (L>>4)*8 + (e&7) + (e>>3)*16.  => two contiguous 16B chunks per lane.
// The same mapping serves B (32x16, K x N) when storage is [N][K] row-major.
// ---------------------------------------------------------------------------
__device__ __forceinline__ v16h load_frag16x32(const _Float16* __restrict__ p, int ld) {
    int lane = threadIdx.x & 31;
    const _Float16* row = p + (lane & 15) * ld + ((lane >> 4) << 3);
    v8h lo = *(const v8h*)(row);
    v8h hi = *(const v8h*)(row + 16);
    v16h f;
#pragma unroll
    for (int i = 0; i < 8; ++i) { f[i] = lo[i]; f[i + 8] = hi[i]; }
    return f;
}

__device__ __forceinline__ v8f zero8() {
    v8f z = {0.f, 0.f, 0.f, 0.f, 0.f, 0.f, 0.f, 0.f};
    return z;
}

// ---------------------------------------------------------------------------
// f32 -> f16 conversion, 4 elements / thread (all sizes divisible by 4)
// ---------------------------------------------------------------------------
__global__ void f32_to_f16_kernel(const float* __restrict__ s, _Float16* __restrict__ d, int n) {
    int i = (blockIdx.x * blockDim.x + threadIdx.x) * 4;
    if (i + 3 < n) {
        float4 v = *(const float4*)(s + i);
        v4h o;
        o[0] = (_Float16)v.x; o[1] = (_Float16)v.y;
        o[2] = (_Float16)v.z; o[3] = (_Float16)v.w;
        *(v4h*)(d + i) = o;
    }
}

#define TLD 40   // padded LDS leading dim (halves); 80B rows keep 16B alignment

// ---------------------------------------------------------------------------
// Shared GEMM core: block tile 256(M) x 64(N), K-step 32, 8 waves / 256 thr.
// Each wave owns a 32x64 strip: 2(M) x 4(N) 16x16 WMMA tiles = 8 WMMA per
// K-step. All 6 fragment loads (2 A + 4 B) are issued before the WMMA chain
// so the dscnt waits retire progressively and the 8 WMMAs issue back-to-back.
// Accumulators acc[2][4] are produced in C-fragment layout:
//   value (row = r + 8*(lane>>4), col = lane&15) in element r.
// ---------------------------------------------------------------------------
template <typename EPILOGUE>
__device__ __forceinline__ void gemm_core_256x64(const _Float16* __restrict__ A,   // [M,1024]
                                                 const _Float16* __restrict__ Bmat,// [N,1024]
                                                 int m0, int n0, EPILOGUE epi) {
    const int K = 1024;
    __shared__ __align__(16) _Float16 As[256 * TLD];
    __shared__ __align__(16) _Float16 Bs[64 * TLD];

    int wave = threadIdx.x >> 5;
    int lane = threadIdx.x & 31;
    int wm = wave * 32;
    int col = lane & 15, hi = lane >> 4;

    v8f acc[2][4];
#pragma unroll
    for (int i = 0; i < 2; ++i)
#pragma unroll
        for (int j = 0; j < 4; ++j) acc[i][j] = zero8();

    for (int k0 = 0; k0 < K; k0 += 32) {
        // ---- stage A 256x32 (1 row / thread, 4x b128) ----
        {
            int row = threadIdx.x;
            const v8h* ga = (const v8h*)(A + (size_t)(m0 + row) * K + k0);
            v8h a0 = ga[0], a1 = ga[1], a2 = ga[2], a3 = ga[3];
            v8h* da = (v8h*)(As + row * TLD);
            da[0] = a0; da[1] = a1; da[2] = a2; da[3] = a3;
        }
        // ---- stage B 64x32 (64 rows x 4 segs, 1x b128 / thread) ----
        {
            int row = threadIdx.x & 63;
            int seg = (threadIdx.x >> 6) * 8;
            v8h b = *(const v8h*)(Bmat + (size_t)(n0 + row) * K + k0 + seg);
            *(v8h*)(Bs + row * TLD + seg) = b;
        }
        __syncthreads();

        // preload ALL fragments, then chain the WMMAs
        v16h a0 = load_frag16x32(As + (wm + 0) * TLD, TLD);
        v16h a1 = load_frag16x32(As + (wm + 16) * TLD, TLD);
        v16h bf[4];
#pragma unroll
        for (int j = 0; j < 4; ++j)
            bf[j] = load_frag16x32(Bs + j * 16 * TLD, TLD);
#pragma unroll
        for (int j = 0; j < 4; ++j) {
            acc[0][j] = WMMA_F16(a0, bf[j], acc[0][j]);
            acc[1][j] = WMMA_F16(a1, bf[j], acc[1][j]);
        }
        __syncthreads();
    }

#pragma unroll
    for (int i = 0; i < 2; ++i)
#pragma unroll
        for (int j = 0; j < 4; ++j) {
            int n = n0 + j * 16 + col;
#pragma unroll
            for (int r = 0; r < 8; ++r) {
                int m = m0 + wm + i * 16 + r + 8 * hi;
                epi(m, n, acc[i][j][r]);
            }
        }
}

// ---------------------------------------------------------------------------
// QKV projection: C[4096,3072] = Xh[4096,1024] @ Wh[3072,1024]^T + bias.
// Scatter-store into q[B,H,S,D] (pre-scaled by d^-1/2), k[B,H,S,D],
// v^T[B,H,D,S] (all f16) for contiguous WMMA fragment loads downstream.
// ---------------------------------------------------------------------------
__global__ void gemm_qkv_kernel(const _Float16* __restrict__ Xh,
                                const _Float16* __restrict__ Wh,
                                const float* __restrict__ bias,
                                _Float16* __restrict__ qh,
                                _Float16* __restrict__ kh,
                                _Float16* __restrict__ vth) {
    int n0 = blockIdx.x * 64;
    int m0 = blockIdx.y * 256;
    gemm_core_256x64(Xh, Wh, m0, n0,
        [=] __device__ (int m, int n, float v) {
            float val = v + bias[n];
            int bb = m >> 11, s = m & 2047;
            if (n < 1024) {
                int hh = n >> 6, d = n & 63;
                qh[(((bb << 4) + hh) * 2048 + s) * 64 + d] = (_Float16)(val * 0.125f);
            } else if (n < 2048) {
                int nn = n - 1024, hh = nn >> 6, d = nn & 63;
                kh[(((bb << 4) + hh) * 2048 + s) * 64 + d] = (_Float16)val;
            } else {
                int nn = n - 2048, hh = nn >> 6, d = nn & 63;
                vth[(((bb << 4) + hh) * 64 + d) * 2048 + s] = (_Float16)val;
            }
        });
}

// ---------------------------------------------------------------------------
// Output projection: out[4096,1024] = Ah[4096,1024] @ Wh[1024,1024]^T + bias
// ---------------------------------------------------------------------------
__global__ void gemm_out_kernel(const _Float16* __restrict__ Ah,
                                const _Float16* __restrict__ Wh,
                                const float* __restrict__ bias,
                                float* __restrict__ Cout) {
    int n0 = blockIdx.x * 64;
    int m0 = blockIdx.y * 256;
    gemm_core_256x64(Ah, Wh, m0, n0,
        [=] __device__ (int m, int n, float v) {
            Cout[(size_t)m * 1024 + n] = v + bias[n];
        });
}

// ---------------------------------------------------------------------------
// Flash-attention with ALiBi. 4 waves/block, each wave owns one 16-row q tile.
// Online softmax over 32-key blocks; P goes via per-wave LDS to convert the
// C-fragment layout into an A-fragment layout for the P @ V^T WMMAs.
// V fragments are fetched from global BEFORE the LDS round-trip so their
// latency hides under softmax + P conversion.
// ---------------------------------------------------------------------------
__global__ void attn_kernel(const _Float16* __restrict__ qh,
                            const _Float16* __restrict__ kh,
                            const _Float16* __restrict__ vth,
                            const unsigned char* __restrict__ mask,
                            _Float16* __restrict__ attnh) {
    const int S = 2048, D = 64, H = 16;
    __shared__ __align__(16) _Float16 Pls[4][16 * TLD];

    int b = blockIdx.z, h = blockIdx.y;
    int wave = threadIdx.x >> 5, lane = threadIdx.x & 31;
    int q0 = blockIdx.x * 64 + wave * 16;
    _Float16* Pl = Pls[wave];

    const _Float16* Q  = qh  + ((size_t)(b * H + h) * S + q0) * D;
    const _Float16* Kp = kh  + (size_t)(b * H + h) * S * D;
    const _Float16* Vt = vth + (size_t)(b * H + h) * D * S;
    const unsigned char* mk = mask + b * S;
    float slope = exp2f(-0.5f * (float)(h + 1));   // exact ALiBi slopes for 16 heads

    int col = lane & 15, hi = lane >> 4;

    v16h qf0 = load_frag16x32(Q, D);          // d = 0..31
    v16h qf1 = load_frag16x32(Q + 32, D);     // d = 32..63

    v8f o[4];
#pragma unroll
    for (int dt = 0; dt < 4; ++dt) o[dt] = zero8();
    float m_i[8], l_i[8];
#pragma unroll
    for (int r = 0; r < 8; ++r) { m_i[r] = -1e30f; l_i[r] = 0.f; }

    for (int n0 = 0; n0 < S; n0 += 32) {
        // --- scores: two 16x16 tiles (keys n0..+15, n0+16..+31), reduce d=64 ---
        v16h k0a = load_frag16x32(Kp + (n0 + 0)  * D, D);
        v16h k0b = load_frag16x32(Kp + (n0 + 0)  * D + 32, D);
        v16h k1a = load_frag16x32(Kp + (n0 + 16) * D, D);
        v16h k1b = load_frag16x32(Kp + (n0 + 16) * D + 32, D);
        v8f s0 = zero8(), s1 = zero8();
        s0 = WMMA_F16(qf0, k0a, s0);
        s0 = WMMA_F16(qf1, k0b, s0);
        s1 = WMMA_F16(qf0, k1a, s1);
        s1 = WMMA_F16(qf1, k1b, s1);

        // --- prefetch V fragments (global, independent of P) ---
        v16h vf[4];
#pragma unroll
        for (int dt = 0; dt < 4; ++dt)
            vf[dt] = load_frag16x32(Vt + (dt * 16) * S + n0, S);

        // --- ALiBi bias + padding mask ---
        int kp0 = n0 + col, kp1 = n0 + 16 + col;
        float mb0 = mk[kp0] ? 0.f : -1e30f;
        float mb1 = mk[kp1] ? 0.f : -1e30f;
#pragma unroll
        for (int r = 0; r < 8; ++r) {
            int qp = q0 + r + 8 * hi;
            s0[r] += mb0 - slope * fabsf((float)(qp - kp0));
            s1[r] += mb1 - slope * fabsf((float)(qp - kp1));
        }

        // --- online softmax (row reductions across 16-lane half-waves) ---
        float alpha[8];
#pragma unroll
        for (int r = 0; r < 8; ++r) {
            float v = fmaxf(s0[r], s1[r]);
            v = fmaxf(v, __shfl_xor(v, 1, 16));
            v = fmaxf(v, __shfl_xor(v, 2, 16));
            v = fmaxf(v, __shfl_xor(v, 4, 16));
            v = fmaxf(v, __shfl_xor(v, 8, 16));
            float mnew = fmaxf(m_i[r], v);
            alpha[r] = __expf(m_i[r] - mnew);
            m_i[r] = mnew;
            s0[r] = __expf(s0[r] - mnew);
            s1[r] = __expf(s1[r] - mnew);
            float rs = s0[r] + s1[r];
            rs += __shfl_xor(rs, 1, 16);
            rs += __shfl_xor(rs, 2, 16);
            rs += __shfl_xor(rs, 4, 16);
            rs += __shfl_xor(rs, 8, 16);
            l_i[r] = l_i[r] * alpha[r] + rs;
        }
#pragma unroll
        for (int dt = 0; dt < 4; ++dt)
#pragma unroll
            for (int r = 0; r < 8; ++r) o[dt][r] *= alpha[r];

        // --- C-frag -> A-frag conversion through per-wave LDS ---
#pragma unroll
        for (int r = 0; r < 8; ++r) {
            Pl[(r + 8 * hi) * TLD + col]      = (_Float16)s0[r];
            Pl[(r + 8 * hi) * TLD + col + 16] = (_Float16)s1[r];
        }
        asm volatile("s_wait_dscnt 0" ::: "memory");
        v16h pf = load_frag16x32(Pl, TLD);

        // --- O += P @ V^T : 4 column tiles of d ---
#pragma unroll
        for (int dt = 0; dt < 4; ++dt)
            o[dt] = WMMA_F16(pf, vf[dt], o[dt]);
    }

    // normalize + store attention output f16 [B*S, H*D]
    float inv[8];
#pragma unroll
    for (int r = 0; r < 8; ++r) inv[r] = (l_i[r] > 0.f) ? (1.f / l_i[r]) : 0.f;
#pragma unroll
    for (int dt = 0; dt < 4; ++dt)
#pragma unroll
        for (int r = 0; r < 8; ++r) {
            int srow = q0 + r + 8 * hi;
            int dcol = h * 64 + dt * 16 + col;
            attnh[(size_t)(b * S + srow) * 1024 + dcol] = (_Float16)(o[dt][r] * inv[r]);
        }
}

// ---------------------------------------------------------------------------
extern "C" void kernel_launch(void* const* d_in, const int* in_sizes, int n_in,
                              void* d_out, int out_size, void* d_ws, size_t ws_size,
                              hipStream_t stream) {
    const float*         x      = (const float*)d_in[0];
    const unsigned char* amask  = (const unsigned char*)d_in[1];
    const float*         qkv_w  = (const float*)d_in[2];
    const float*         qkv_b  = (const float*)d_in[3];
    const float*         out_w  = (const float*)d_in[4];
    const float*         out_b  = (const float*)d_in[5];
    float*               out    = (float*)d_out;

    char* w = (char*)d_ws;
    _Float16* xh    = (_Float16*)(w);                          // 8 MB
    _Float16* wqkvh = (_Float16*)(w + (size_t)8  * 1048576);   // 6 MB
    _Float16* wouth = (_Float16*)(w + (size_t)14 * 1048576);   // 2 MB
    _Float16* qh    = (_Float16*)(w + (size_t)16 * 1048576);   // 8 MB
    _Float16* kh    = (_Float16*)(w + (size_t)24 * 1048576);   // 8 MB
    _Float16* vth   = (_Float16*)(w + (size_t)32 * 1048576);   // 8 MB
    _Float16* ah    = (_Float16*)(w + (size_t)40 * 1048576);   // 8 MB

    const int NX  = 2 * 2048 * 1024;      // 4,194,304
    const int NW1 = 3072 * 1024;          // 3,145,728
    const int NW2 = 1024 * 1024;          // 1,048,576

    f32_to_f16_kernel<<<NX  / 4 / 256, 256, 0, stream>>>(x,     xh,    NX);
    f32_to_f16_kernel<<<NW1 / 4 / 256, 256, 0, stream>>>(qkv_w, wqkvh, NW1);
    f32_to_f16_kernel<<<NW2 / 4 / 256, 256, 0, stream>>>(out_w, wouth, NW2);

    gemm_qkv_kernel<<<dim3(3072 / 64, 4096 / 256), 256, 0, stream>>>(
        xh, wqkvh, qkv_b, qh, kh, vth);

    attn_kernel<<<dim3(2048 / 64, 16, 2), 128, 0, stream>>>(
        qh, kh, vth, amask, ah);

    gemm_out_kernel<<<dim3(1024 / 64, 4096 / 256), 256, 0, stream>>>(
        ah, wouth, out_b, out);
}
